// CycleWiseSelfAttention_69260642615588
// MI455X (gfx1250) — matrix operand order repaired and checked by reading, not compiled, not run on hardware
//
#include <hip/hip_runtime.h>
#include <stdint.h>

// ---------------------------------------------------------------------------
// CycleWiseSelfAttention for MI455X (gfx1250, wave32, WMMA bf16).
//   B=8, C=16, S=512, E=256.
//   Pass 1: per-cycle projection GEMMs (fp32 -> bf16 WMMA, f32 accum),
//           bias + ReLU (+ 1/sqrt(E) folded into Q). V stored transposed.
//   Pass 2: flash-style attention; K/V chunks streamed into double-buffered
//           LDS with GLOBAL_LOAD_ASYNC_TO_LDS_B128 (ASYNCcnt), WMMAs overlap
//           the DMA of the next chunk.
// ds_read<->WMMA interleave enforced with sched_group_barrier so each WMMA
// waits dscnt<=2 with the next fragment's loads already in flight.
// Workspace: Qb/Kb (bf16 [BC][S][E]) + VTb (bf16 [BC][E][S]) = 100.7 MB.
// ---------------------------------------------------------------------------

typedef __attribute__((ext_vector_type(16))) __bf16 v16bf;
typedef __attribute__((ext_vector_type(8)))  float  v8f;
typedef int v4i_t __attribute__((vector_size(16)));

union Frag16 { uint32_t u[8]; v16bf v; };

constexpr int Bn = 8, Cn = 16, Sn = 512, En = 256, BC = Bn * Cn;

#define GPTR(p) ((__attribute__((address_space(1))) v4i_t*)(p))
#define LPTR(p) ((__attribute__((address_space(3))) v4i_t*)(p))

__device__ __forceinline__ uint16_t f2bf(float x) {
  uint32_t u = __float_as_uint(x);
  u += 0x7FFFu + ((u >> 16) & 1u);           // round-to-nearest-even
  return (uint16_t)(u >> 16);
}
__device__ __forceinline__ uint32_t pack2(uint16_t lo, uint16_t hi) {
  return (uint32_t)lo | ((uint32_t)hi << 16);
}

// Scheduling groups: 0x100 = DS read, 0x008 = MFMA/WMMA.
// Distance-1 software pipeline over 16 WMMAs whose fragments are 2 ds reads
// each; PRE = ds reads issued before the first WMMA (A-frag + 2 buffers).
template <int PRE>
__device__ __forceinline__ void sched_pipeline16() {
  __builtin_amdgcn_sched_group_barrier(0x100, PRE, 0);
  __builtin_amdgcn_sched_group_barrier(0x008, 1, 0);
  #pragma unroll
  for (int i = 0; i < 14; ++i) {
    __builtin_amdgcn_sched_group_barrier(0x100, 2, 0);
    __builtin_amdgcn_sched_group_barrier(0x008, 1, 0);
  }
  __builtin_amdgcn_sched_group_barrier(0x008, 1, 0);
}

// B-matrix fragment (32x16, K x N), same K order as A: VGPR 0..3 hold
// K={0..7}+8*half, VGPR 4..7 hold K={16..23}+8*half, for column N = lane&15.
// rowbase points at 32 consecutive K-elements of one column.  16B-aligned b128 reads.
__device__ __forceinline__ void load_bfrag(Frag16& F, const uint16_t* rowbase, int h) {
  uint4 lo = *(const uint4*)(rowbase + 8 * h);
  uint4 hi = *(const uint4*)(rowbase + 16 + 8 * h);
  F.u[0] = lo.x; F.u[1] = lo.y; F.u[2] = lo.z; F.u[3] = lo.w;
  F.u[4] = hi.x; F.u[5] = hi.y; F.u[6] = hi.z; F.u[7] = hi.w;
}
// A-matrix fragment (16x32, M x K), ISA layout: lane m = M row; VGPR 0..3 hold
// K={0..7}+8*half, VGPR 4..7 hold K={16..23}+8*half.
__device__ __forceinline__ void load_afrag(Frag16& F, const uint16_t* rowbase, int h) {
  uint4 lo = *(const uint4*)(rowbase + 8 * h);
  uint4 hi = *(const uint4*)(rowbase + 16 + 8 * h);
  F.u[0] = lo.x; F.u[1] = lo.y; F.u[2] = lo.z; F.u[3] = lo.w;
  F.u[4] = hi.x; F.u[5] = hi.y; F.u[6] = hi.z; F.u[7] = hi.w;
}
__device__ __forceinline__ v8f wmma_bf16(const Frag16& A, const Frag16& B, v8f C) {
  return __builtin_amdgcn_wmma_f32_16x16x32_bf16(false, A.v, false, B.v,
                                                 (short)0, C, false, false);
}

// ---------------------------------------------------------------------------
// Projection: out[bc] = relu(X[bc] @ W[c] + bias[c]) * scale  (bf16 output)
// grid = BC * (S/128) blocks, 256 threads (8 waves x 16 rows).
// ---------------------------------------------------------------------------
__global__ __launch_bounds__(256) void proj_kernel(
    const float* __restrict__ X, const float* __restrict__ W,
    const float* __restrict__ bias, uint16_t* __restrict__ out,
    float scale, int transposeOut)
{
  __shared__ alignas(16) uint16_t Xs[128][40];      // bf16 X slab (converted at staging)
  __shared__ alignas(16) uint16_t Wt[256][40];      // bf16 W^T slab [f][e-chunk]
  __shared__ alignas(16) uint16_t stg[8][16][16];   // per-wave store-staging tile

  const int tid  = threadIdx.x;
  const int wave = tid >> 5;
  const int lane = tid & 31;
  const int m    = lane & 15;      // A row / B-C-D column
  const int h    = lane >> 4;      // lane half
  const int bc   = blockIdx.x >> 2;
  const int sblk = blockIdx.x & 3;
  const int cyc  = bc & (Cn - 1);
  const int s0   = sblk * 128;
  const int srow = s0 + wave * 16;

  const float* Xbc = X + (size_t)bc  * Sn * En;
  const float* Wc  = W + (size_t)cyc * En * En;

  const int xrow = tid >> 1, xc0 = (tid & 1) * 16;  // X staging role
  const int we   = tid >> 3, wf0 = (tid & 7) * 32;  // W staging role

  float4 Xr[4], Wr[8];                              // register-staged next chunk
  auto load_slabs = [&](int k) {
    const int e0 = k * 32;
    const float* xr = Xbc + (size_t)(s0 + xrow) * En + e0 + xc0;
    #pragma unroll
    for (int i = 0; i < 4; ++i) Xr[i] = *(const float4*)(xr + i * 4);
    const float* wr = Wc + (size_t)(e0 + we) * En + wf0;
    #pragma unroll
    for (int i = 0; i < 8; ++i) Wr[i] = *(const float4*)(wr + i * 4);
  };

  v8f acc[16];
  #pragma unroll
  for (int i = 0; i < 16; ++i)
    #pragma unroll
    for (int r = 0; r < 8; ++r) acc[i][r] = 0.0f;

  load_slabs(0);
  for (int k = 0; k < 8; ++k) {                     // e-chunks of 32
    __syncthreads();                                // prev compute done reading LDS
    {   // commit staged X slab as packed bf16 (two b128 DS stores)
      uint32_t p[8];
      #pragma unroll
      for (int i = 0; i < 4; ++i) {
        p[2*i]   = pack2(f2bf(Xr[i].x), f2bf(Xr[i].y));
        p[2*i+1] = pack2(f2bf(Xr[i].z), f2bf(Xr[i].w));
      }
      *(uint4*)&Xs[xrow][xc0]     = make_uint4(p[0], p[1], p[2], p[3]);
      *(uint4*)&Xs[xrow][xc0 + 8] = make_uint4(p[4], p[5], p[6], p[7]);
    }
    {   // commit staged W^T slab (transpose scatter, bf16)
      #pragma unroll
      for (int i = 0; i < 8; ++i) {
        Wt[wf0 + i*4 + 0][we] = f2bf(Wr[i].x);
        Wt[wf0 + i*4 + 1][we] = f2bf(Wr[i].y);
        Wt[wf0 + i*4 + 2][we] = f2bf(Wr[i].z);
        Wt[wf0 + i*4 + 3][we] = f2bf(Wr[i].w);
      }
    }
    __syncthreads();
    load_slabs((k + 1) & 7);      // branchless: last iter reloads chunk 0 (discarded)

    Frag16 A;
    load_afrag(A, &Xs[wave * 16 + m][0], h);

    Frag16 Bf[2];                                   // double-buffered B fragments
    load_bfrag(Bf[0], &Wt[m][0], h);
    #pragma unroll
    for (int jf = 0; jf < 16; ++jf) {
      if (jf < 15) load_bfrag(Bf[(jf + 1) & 1], &Wt[(jf + 1) * 16 + m][0], h);
      acc[jf] = wmma_bf16(A, Bf[jf & 1], acc[jf]);
    }
    sched_pipeline16<6>();        // A(2)+B0(2)+B1(2) ahead of first WMMA
  }

  // Epilogue: bias + ReLU + scale -> bf16, coalesce via per-wave LDS tile.
  const int tr = lane >> 1, tc = (lane & 1) * 8;
  #pragma unroll
  for (int jf = 0; jf < 16; ++jf) {
    float bn = bias[(size_t)cyc * En + jf * 16 + m];
    #pragma unroll
    for (int r = 0; r < 8; ++r) {
      float val = fmaxf(acc[jf][r] + bn, 0.0f) * scale;
      int row = r + 8 * h;                          // C/D layout: row r+8*half
      if (transposeOut) stg[wave][m][row] = f2bf(val);
      else              stg[wave][row][m] = f2bf(val);
    }
    // wave-internal DS ops are in-order: safe to read back without barrier
    uint4 blk = *(const uint4*)&stg[wave][tr][tc];
    if (transposeOut) {  // V^T: [E][S]
      uint16_t* dst = out + (size_t)bc * En * Sn + (size_t)(jf * 16 + tr) * Sn + srow + tc;
      *(uint4*)dst = blk;
    } else {             // Q/K: [S][E]
      uint16_t* dst = out + (size_t)bc * Sn * En + (size_t)(srow + tr) * En + jf * 16 + tc;
      *(uint4*)dst = blk;
    }
  }
}

// ---------------------------------------------------------------------------
// Flash attention: softmax(Q K^T) V, one pass over keys in chunks of 32.
// grid = BC * (S/128) blocks, 256 threads; each wave owns 16 query rows.
// K/V chunks DMA'd into double-buffered LDS via async-to-LDS (ASYNCcnt).
// ---------------------------------------------------------------------------
__global__ __launch_bounds__(256) void attn_kernel(
    const uint16_t* __restrict__ Qb, const uint16_t* __restrict__ Kb,
    const uint16_t* __restrict__ VTb, float* __restrict__ out)
{
  __shared__ alignas(16) uint16_t Ks [2][32][264];  // K chunk  [t][e], pad 8
  __shared__ alignas(16) uint16_t Vts[2][256][40];  // V^T chunk [e][t], pad 8
  __shared__ alignas(16) uint16_t Ps [8][16][40];   // per-wave P relayout tile

  const int tid  = threadIdx.x;
  const int wave = tid >> 5;
  const int lane = tid & 31;
  const int m    = lane & 15;
  const int h    = lane >> 4;
  const int bc   = blockIdx.x >> 2;
  const int sblk = blockIdx.x & 3;
  const int srow = sblk * 128 + wave * 16;

  const int krow = tid >> 3, kseg = tid & 7;
  const uint16_t* Kbase = Kb  + (size_t)bc * Sn * En;
  const uint16_t* Vbase = VTb + (size_t)bc * En * Sn + (size_t)tid * Sn;

  // Async DMA of one 32-key chunk into LDS buffer `buf` (8 b128 per lane).
  auto async_chunk = [&](int t0, int buf) {
    const uint16_t* kr = Kbase + (size_t)(t0 + krow) * En + kseg * 32;
    uint16_t* kd = &Ks[buf][krow][kseg * 32];
    #pragma unroll
    for (int i = 0; i < 4; ++i)
      __builtin_amdgcn_global_load_async_to_lds_b128(GPTR(kr + i * 8), LPTR(kd + i * 8), 0, 0);
    const uint16_t* vr = Vbase + t0;
    uint16_t* vd = &Vts[buf][tid][0];
    #pragma unroll
    for (int i = 0; i < 4; ++i)
      __builtin_amdgcn_global_load_async_to_lds_b128(GPTR(vr + i * 8), LPTR(vd + i * 8), 0, 0);
  };

  // Q A-fragments for all 8 e-chunks, resident in registers (64 VGPRs).
  Frag16 Aq[8];
  {
    const uint16_t* qr = Qb + (size_t)bc * Sn * En + (size_t)(srow + m) * En;
    #pragma unroll
    for (int k = 0; k < 8; ++k) load_afrag(Aq[k], qr + k * 32, h);
  }

  float mi[8], li[8];
  #pragma unroll
  for (int r = 0; r < 8; ++r) { mi[r] = -3.0e38f; li[r] = 0.0f; }
  v8f acc[16];
  #pragma unroll
  for (int i = 0; i < 16; ++i)
    #pragma unroll
    for (int r = 0; r < 8; ++r) acc[i][r] = 0.0f;

  async_chunk(0, 0);                              // prologue: fill buffer 0
  for (int tcn = 0; tcn < 16; ++tcn) {            // key chunks of 32
    // Wait for this chunk's DMA (per-wave), then sync all waves' DMAs.
    asm volatile("s_wait_asynccnt 0" ::: "memory");
    __syncthreads();
    const int buf = tcn & 1;
    // Kick off DMA of the next chunk into the other buffer (wrap on last).
    const int t0n = ((tcn + 1) & 15) * 32;
    async_chunk(t0n, buf ^ 1);
    // Warm L2 two chunks ahead.
    const int t0p = ((tcn + 2) & 15) * 32;
    __builtin_prefetch(Kbase + (size_t)(t0p + krow) * En + kseg * 32, 0, 1);
    __builtin_prefetch(Vbase + t0p, 0, 1);

    // Scores S = Q K^T for 16 rows x 32 cols (two 16x16 C tiles),
    // B fragments double-buffered across the flattened (k, jt) loop.
    v8f sc[2];
    #pragma unroll
    for (int jt = 0; jt < 2; ++jt)
      #pragma unroll
      for (int r = 0; r < 8; ++r) sc[jt][r] = 0.0f;
    {
      Frag16 Bf[2];
      load_bfrag(Bf[0], &Ks[buf][m][0], h);       // (k=0, jt=0)
      #pragma unroll
      for (int q = 0; q < 16; ++q) {
        const int k = q >> 1, jt = q & 1;
        if (q < 15) {
          const int kn = (q + 1) >> 1, jtn = (q + 1) & 1;
          load_bfrag(Bf[(q + 1) & 1], &Ks[buf][jtn * 16 + m][kn * 32], h);
        }
        sc[jt] = wmma_bf16(Aq[k], Bf[q & 1], sc[jt]);
      }
      sched_pipeline16<4>();      // B0(2)+B1(2) ahead of first WMMA
    }

    // Online softmax per row (row r+8h lives in lanes of half h, col = lane&15).
    #pragma unroll
    for (int r = 0; r < 8; ++r) {
      float v0 = sc[0][r], v1 = sc[1][r];
      float vm = fmaxf(v0, v1);
      #pragma unroll
      for (int d = 1; d < 16; d <<= 1) vm = fmaxf(vm, __shfl_xor(vm, d, 32));
      float mnew  = fmaxf(mi[r], vm);
      float alpha = __expf(mi[r] - mnew);
      float p0 = __expf(v0 - mnew);
      float p1 = __expf(v1 - mnew);
      float rs = p0 + p1;
      #pragma unroll
      for (int d = 1; d < 16; d <<= 1) rs += __shfl_xor(rs, d, 32);
      li[r] = li[r] * alpha + rs;
      mi[r] = mnew;
      #pragma unroll
      for (int jf = 0; jf < 16; ++jf) acc[jf][r] *= alpha;   // rescale O
      int row = r + 8 * h;
      Ps[wave][row][m]      = f2bf(p0);           // relayout C/D -> A via LDS
      Ps[wave][row][16 + m] = f2bf(p1);
    }

    // P (16x32) as A-fragment; wave-internal DS ordering covers the RAW.
    Frag16 Pa;
    load_afrag(Pa, &Ps[wave][m][0], h);

    // O += P V  (V^T chunk gives contiguous B-fragments), double-buffered.
    {
      Frag16 Bv[2];
      load_bfrag(Bv[0], &Vts[buf][m][0], h);
      #pragma unroll
      for (int jf = 0; jf < 16; ++jf) {
        if (jf < 15) load_bfrag(Bv[(jf + 1) & 1], &Vts[buf][(jf + 1) * 16 + m][0], h);
        acc[jf] = wmma_bf16(Pa, Bv[jf & 1], acc[jf]);
      }
      sched_pipeline16<6>();      // Pa(2)+Bv0(2)+Bv1(2) ahead of first WMMA
    }
  }

  // Normalize and store fp32 output [B][C][S][E].
  float* obase = out + (size_t)bc * Sn * En;
  #pragma unroll
  for (int r = 0; r < 8; ++r) {
    float inv = 1.0f / li[r];
    float* orow = obase + (size_t)(srow + r + 8 * h) * En;
    #pragma unroll
    for (int jf = 0; jf < 16; ++jf) orow[jf * 16 + m] = acc[jf][r] * inv;
  }
}

// ---------------------------------------------------------------------------
extern "C" void kernel_launch(void* const* d_in, const int* in_sizes, int n_in,
                              void* d_out, int out_size, void* d_ws, size_t ws_size,
                              hipStream_t stream) {
  const float* query = (const float*)d_in[0];
  const float* key_  = (const float*)d_in[1];
  const float* value = (const float*)d_in[2];
  const float* Wq    = (const float*)d_in[3];
  const float* Wk    = (const float*)d_in[4];
  const float* Wv    = (const float*)d_in[5];
  const float* bq    = (const float*)d_in[6];
  const float* bk    = (const float*)d_in[7];
  const float* bv    = (const float*)d_in[8];
  float* out = (float*)d_out;

  // Workspace: Qb, Kb bf16 [BC][S][E]; VTb bf16 [BC][E][S]  (3 x 33.6 MB).
  uint16_t* Qb  = (uint16_t*)d_ws;
  uint16_t* Kb  = Qb + (size_t)BC * Sn * En;
  uint16_t* VTb = Kb + (size_t)BC * Sn * En;

  const float qscale = 0.0625f;   // E^-0.5 = 1/16, folded into Q projection
  dim3 blk(256);
  dim3 grd(BC * (Sn / 128));      // 512 blocks

  proj_kernel<<<grd, blk, 0, stream>>>(query, Wq, bq, Qb,  qscale, 0);
  proj_kernel<<<grd, blk, 0, stream>>>(key_,  Wk, bk, Kb,  1.0f,   0);
  proj_kernel<<<grd, blk, 0, stream>>>(value, Wv, bv, VTb, 1.0f,   1);
  attn_kernel<<<grd, blk, 0, stream>>>(Qb, Kb, VTb, out);
}
